// EncoderLayer_39865886442145
// MI455X (gfx1250) — compile-verified, matrix operand order checked by reading, not run on hardware
//
#include <hip/hip_runtime.h>

// ---------------------------------------------------------------------------
// Transformer encoder layer for MI455X (gfx1250, wave32, WMMA + TDM).
//   B=8, N=1024, E=512, H=8, D=64, F=2048  -> M = B*N = 8192 token rows.
// GEMMs: v_wmma_f32_16x16x32_bf16, LDS tiles staged by the Tensor Data Mover
// (tensor_load_to_lds + s_wait_tensorcnt).  Attention is flash-style.
// ---------------------------------------------------------------------------

#define B_ 8
#define N_ 1024
#define E_ 512
#define H_ 8
#define D_ 64
#define F_ 2048
#define M_ (B_ * N_)   // 8192

typedef __attribute__((ext_vector_type(16))) __bf16 v16bf;
typedef __attribute__((ext_vector_type(8)))  float  v8f;
typedef unsigned int v4u __attribute__((ext_vector_type(4)));
typedef unsigned int v8u __attribute__((ext_vector_type(8)));

// float -> bf16 bits, round-to-nearest-even
__device__ __forceinline__ unsigned short f2bf_bits(float f) {
    union { float f; unsigned int u; } cv; cv.f = f;
    unsigned int u = cv.u;
    unsigned int r = (u + 0x7FFFu + ((u >> 16) & 1u)) >> 16;
    return (unsigned short)r;
}
__device__ __forceinline__ __bf16 bits2bf(unsigned short s) {
    union { unsigned short s; __bf16 h; } o; o.s = s; return o.h;
}

// ---------------------------------------------------------------------------
// Tensor Data Mover: DMA a 2D tile (16-bit elements) Global -> LDS.
// Builds the D# (CDNA5 ISA) in SGPRs and issues
// tensor_load_to_lds via inline asm (portable across both toolchains).
//   group0: [1:0]=count=1 | [63:32]=lds_addr | [120:64]=global_addr |
//           [127:126]=type=2
//   group1: [17:16]=data_size(1->2B) | tensor_dim0/1 | tile_dim0/1 |
//           tensor_dim0_stride (elements)
// All arguments must be wave-uniform (scalar).
// ---------------------------------------------------------------------------
__device__ __forceinline__ void tdm_load_2d_bf16(
        unsigned lds_addr, const unsigned short* gptr,
        unsigned tile_d0, unsigned tile_d1, unsigned stride0_elems) {
    unsigned long long ga = (unsigned long long)(size_t)gptr;
    v4u g0;
    g0[0] = 1u;                                         // count = 1 (valid D#)
    g0[1] = lds_addr;                                   // LDS byte address
    g0[2] = (unsigned)(ga & 0xFFFFFFFFu);               // global_addr[31:0]
    g0[3] = (unsigned)((ga >> 32) & 0x01FFFFFFu)        // global_addr[56:32]
            | (2u << 30);                               // type = 2 ("image")
    v8u g1;
    g1[0] = 1u << 16;                                   // data_size=1 (2B), mask=0
    g1[1] = (tile_d0 & 0xFFFFu) << 16;                  // tensor_dim0 = tile_d0
    g1[2] = (tile_d0 >> 16) | ((tile_d1 & 0xFFFFu) << 16); // tensor_dim1 = tile_d1
    g1[3] = (tile_d1 >> 16) | ((tile_d0 & 0xFFFFu) << 16); // tile_dim0
    g1[4] = tile_d1 & 0xFFFFu;                          // tile_dim1 (tile_dim2=0)
    g1[5] = stride0_elems;                              // tensor_dim0_stride[31:0]
    g1[6] = 0u;
    g1[7] = 0u;
    asm volatile("tensor_load_to_lds %0, %1" :: "s"(g0), "s"(g1) : "memory");
}

// ---------------------------------------------------------------------------
// fp32 -> bf16 conversion (grid-stride)
// ---------------------------------------------------------------------------
__global__ __launch_bounds__(256) void convert_bf16_kernel(
        const float* __restrict__ src, unsigned short* __restrict__ dst, int n) {
    for (int i = blockIdx.x * blockDim.x + threadIdx.x; i < n;
         i += gridDim.x * blockDim.x)
        dst[i] = f2bf_bits(src[i]);
}

// ---------------------------------------------------------------------------
// bf16 WMMA GEMM:  C[M,N] = A[M,K] * B[K,N] (+bias) (ReLU) -> f32/bf16
// Block = 256 threads = 8 waves, macro-tile 64(M) x 128(N), K-step 32.
// Wave w: 16-row sub-tile tr = w&3, 64-col strip tc = w>>2 -> 4 accumulators;
// one A-operand gather feeds 4 WMMAs per K-step.
// Tile staging: per-wave TDM (wave w DMAs A rows [8w,8w+8), B rows [4w,4w+4)),
// then s_wait_tensorcnt 0 + barrier.
// Operand gather follows CDNA5 ISA 16-bit A/B VGPR layouts:
//   element i (i=2v+p):  k = (v<4?0:16) + (lane/16)*8 + (v&3)*2 + p
//   A: row = lane%16 ; B: col = lane%16
// C layout: vgpr r -> row r + 8*(lane/16), col = lane%16
// ---------------------------------------------------------------------------
template <bool BIAS, bool RELU, bool BF16OUT>
__global__ __launch_bounds__(256) void gemm_bf16_kernel(
        const unsigned short* __restrict__ Ag,
        const unsigned short* __restrict__ Bg,
        const float* __restrict__ bias,
        float* __restrict__ outF, unsigned short* __restrict__ outB,
        int M, int Nn, int K) {
    __shared__ unsigned short As[64][32];    // 4 KB
    __shared__ unsigned short Bs[32][128];   // 8 KB

    const int tid  = threadIdx.x;
    const int wv   = __builtin_amdgcn_readfirstlane(tid >> 5);  // scalar wave id
    const int lane = tid & 31;
    const int half = lane >> 4;
    const int l16  = lane & 15;
    const int m0 = blockIdx.y * 64;
    const int n0 = blockIdx.x * 128;
    const int tr = wv & 3;   // M sub-tile (16 rows)
    const int tc = wv >> 2;  // N strip (64 cols)

    const unsigned ldsA = (unsigned)(size_t)&As[0][0];
    const unsigned ldsB = (unsigned)(size_t)&Bs[0][0];

    v8f c[4] = {v8f{}, v8f{}, v8f{}, v8f{}};

    for (int k0 = 0; k0 < K; k0 += 32) {
        // --- TDM staging: each wave DMAs its slice of the A and B tiles ---
        tdm_load_2d_bf16(ldsA + (unsigned)(wv * 8 * 32 * 2),
                         &Ag[(size_t)(m0 + wv * 8) * K + k0],
                         /*tile_d0=*/32, /*tile_d1=*/8, /*stride=*/(unsigned)K);
        tdm_load_2d_bf16(ldsB + (unsigned)(wv * 4 * 128 * 2),
                         &Bg[(size_t)(k0 + wv * 4) * Nn + n0],
                         /*tile_d0=*/128, /*tile_d1=*/4, /*stride=*/(unsigned)Nn);
        __builtin_amdgcn_s_wait_tensorcnt(0);
        __syncthreads();

        v16bf a;
#pragma unroll
        for (int i = 0; i < 16; ++i) {
            const int v = i >> 1, p = i & 1;
            const int kk = ((v < 4) ? 0 : 16) + half * 8 + (v & 3) * 2 + p;
            a[i] = bits2bf(As[tr * 16 + l16][kk]);
        }
#pragma unroll
        for (int ct = 0; ct < 4; ++ct) {
            v16bf bv;
#pragma unroll
            for (int i = 0; i < 16; ++i) {
                const int v = i >> 1, p = i & 1;
                const int kk = ((v < 4) ? 0 : 16) + half * 8 + (v & 3) * 2 + p;
                bv[i] = bits2bf(Bs[kk][tc * 64 + ct * 16 + l16]);
            }
            c[ct] = __builtin_amdgcn_wmma_f32_16x16x32_bf16(false, a, false, bv,
                                                            (short)0, c[ct],
                                                            false, false);
        }
        __syncthreads();
    }

#pragma unroll
    for (int ct = 0; ct < 4; ++ct) {
#pragma unroll
        for (int r = 0; r < 8; ++r) {
            const int gm = m0 + tr * 16 + r + 8 * half;
            const int gn = n0 + tc * 64 + ct * 16 + l16;
            float x = c[ct][r];
            if (BIAS) x += bias[gn];
            if (RELU) x = fmaxf(x, 0.0f);
            if (BF16OUT) outB[(size_t)gm * Nn + gn] = f2bf_bits(x);
            else         outF[(size_t)gm * Nn + gn] = x;
        }
    }
}

// ---------------------------------------------------------------------------
// Flash attention: one wave per (b, h, 16-query tile).  (unchanged)
// ---------------------------------------------------------------------------
__global__ __launch_bounds__(32) void flash_attn_kernel(
        const unsigned short* __restrict__ Qg,
        const unsigned short* __restrict__ Kg,
        const unsigned short* __restrict__ Vg,
        unsigned short* __restrict__ Og) {
    __shared__ unsigned short Qs[16][64];
    __shared__ unsigned short Ks[32][64];
    __shared__ unsigned short Vs[32][64];
    __shared__ unsigned short Ps[16][32];

    const int lane = threadIdx.x;
    const int half = lane >> 4;
    const int l16  = lane & 15;
    const int q0 = blockIdx.x * 16;
    const int b  = blockIdx.y / H_;
    const int h  = blockIdx.y % H_;
    const size_t rowstride = H_ * D_;  // 512
    const size_t base = ((size_t)b * N_) * rowstride + (size_t)h * D_;

#pragma unroll
    for (int t = 0; t < 4; ++t) {
        int u = lane * 4 + t;
        int r = u >> 3, d = (u & 7) * 8;
        *reinterpret_cast<uint4*>(&Qs[r][d]) =
            *reinterpret_cast<const uint4*>(&Qg[base + (size_t)(q0 + r) * rowstride + d]);
    }
    __syncthreads();

    v16bf aq0, aq1;
#pragma unroll
    for (int i = 0; i < 16; ++i) {
        const int v = i >> 1, p = i & 1;
        const int kk = ((v < 4) ? 0 : 16) + half * 8 + (v & 3) * 2 + p;
        aq0[i] = bits2bf(Qs[l16][kk]);
        aq1[i] = bits2bf(Qs[l16][32 + kk]);
    }

    float m_r[8], l_r[8];
    v8f acc[4] = {v8f{}, v8f{}, v8f{}, v8f{}};
#pragma unroll
    for (int r = 0; r < 8; ++r) { m_r[r] = -3.0e38f; l_r[r] = 0.0f; }

    const float scale = 0.125f;  // 1/sqrt(64)

    for (int kt = 0; kt < N_ / 32; ++kt) {
        const int kb = kt * 32;
#pragma unroll
        for (int t = 0; t < 8; ++t) {
            int d = t * 8;
            *reinterpret_cast<uint4*>(&Ks[lane][d]) =
                *reinterpret_cast<const uint4*>(&Kg[base + (size_t)(kb + lane) * rowstride + d]);
            *reinterpret_cast<uint4*>(&Vs[lane][d]) =
                *reinterpret_cast<const uint4*>(&Vg[base + (size_t)(kb + lane) * rowstride + d]);
        }
        __syncthreads();

        v8f s0 = {}, s1 = {};
        {
            v16bf bk;
#pragma unroll
            for (int i = 0; i < 16; ++i) {
                const int v = i >> 1, p = i & 1;
                const int kk = ((v < 4) ? 0 : 16) + half * 8 + (v & 3) * 2 + p;
                bk[i] = bits2bf(Ks[l16][kk]);
            }
            s0 = __builtin_amdgcn_wmma_f32_16x16x32_bf16(false, aq0, false, bk,
                                                         (short)0, s0, false, false);
#pragma unroll
            for (int i = 0; i < 16; ++i) {
                const int v = i >> 1, p = i & 1;
                const int kk = ((v < 4) ? 0 : 16) + half * 8 + (v & 3) * 2 + p;
                bk[i] = bits2bf(Ks[l16][32 + kk]);
            }
            s0 = __builtin_amdgcn_wmma_f32_16x16x32_bf16(false, aq1, false, bk,
                                                         (short)0, s0, false, false);
#pragma unroll
            for (int i = 0; i < 16; ++i) {
                const int v = i >> 1, p = i & 1;
                const int kk = ((v < 4) ? 0 : 16) + half * 8 + (v & 3) * 2 + p;
                bk[i] = bits2bf(Ks[16 + l16][kk]);
            }
            s1 = __builtin_amdgcn_wmma_f32_16x16x32_bf16(false, aq0, false, bk,
                                                         (short)0, s1, false, false);
#pragma unroll
            for (int i = 0; i < 16; ++i) {
                const int v = i >> 1, p = i & 1;
                const int kk = ((v < 4) ? 0 : 16) + half * 8 + (v & 3) * 2 + p;
                bk[i] = bits2bf(Ks[16 + l16][32 + kk]);
            }
            s1 = __builtin_amdgcn_wmma_f32_16x16x32_bf16(false, aq1, false, bk,
                                                         (short)0, s1, false, false);
        }

#pragma unroll
        for (int r = 0; r < 8; ++r) {
            float v0 = s0[r] * scale, v1 = s1[r] * scale;
            float mx = fmaxf(v0, v1);
            mx = fmaxf(mx, __shfl_xor(mx, 1, 32));
            mx = fmaxf(mx, __shfl_xor(mx, 2, 32));
            mx = fmaxf(mx, __shfl_xor(mx, 4, 32));
            mx = fmaxf(mx, __shfl_xor(mx, 8, 32));
            const float nm    = fmaxf(m_r[r], mx);
            const float alpha = __expf(m_r[r] - nm);
            const float p0 = __expf(v0 - nm);
            const float p1 = __expf(v1 - nm);
            m_r[r] = nm;
            float ps = p0 + p1;
            ps += __shfl_xor(ps, 1, 32);
            ps += __shfl_xor(ps, 2, 32);
            ps += __shfl_xor(ps, 4, 32);
            ps += __shfl_xor(ps, 8, 32);
            l_r[r] = l_r[r] * alpha + ps;
#pragma unroll
            for (int ct = 0; ct < 4; ++ct) acc[ct][r] *= alpha;
            Ps[r + 8 * half][l16]      = f2bf_bits(p0);
            Ps[r + 8 * half][16 + l16] = f2bf_bits(p1);
        }
        __syncthreads();

        v16bf ap;
#pragma unroll
        for (int i = 0; i < 16; ++i) {
            const int v = i >> 1, p = i & 1;
            const int kk = ((v < 4) ? 0 : 16) + half * 8 + (v & 3) * 2 + p;
            ap[i] = bits2bf(Ps[l16][kk]);
        }
#pragma unroll
        for (int ct = 0; ct < 4; ++ct) {
            v16bf bv;
#pragma unroll
            for (int i = 0; i < 16; ++i) {
                const int v = i >> 1, p = i & 1;
                const int kk = ((v < 4) ? 0 : 16) + half * 8 + (v & 3) * 2 + p;
                bv[i] = bits2bf(Vs[kk][ct * 16 + l16]);
            }
            acc[ct] = __builtin_amdgcn_wmma_f32_16x16x32_bf16(false, ap, false, bv,
                                                              (short)0, acc[ct],
                                                              false, false);
        }
        __syncthreads();
    }

#pragma unroll
    for (int ct = 0; ct < 4; ++ct) {
#pragma unroll
        for (int r = 0; r < 8; ++r) {
            const float inv = 1.0f / l_r[r];
            const int row = q0 + r + 8 * half;
            const int d   = ct * 16 + l16;
            Og[base + (size_t)row * rowstride + d] = f2bf_bits(acc[ct][r] * inv);
        }
    }
}

// ---------------------------------------------------------------------------
// out = LayerNorm(Ax + Bx) * g + b ; optional bf16 shadow copy for next GEMM.
// ---------------------------------------------------------------------------
__global__ __launch_bounds__(256) void add_layernorm_kernel(
        const float* __restrict__ Ax, const float* __restrict__ Bx,
        const float* __restrict__ g,  const float* __restrict__ be,
        float* __restrict__ outF, unsigned short* __restrict__ outB) {
    __shared__ float red[256];
    const int row = blockIdx.x, t = threadIdx.x;
    const size_t base = (size_t)row * E_;

    const float v0 = Ax[base + t]       + Bx[base + t];
    const float v1 = Ax[base + 256 + t] + Bx[base + 256 + t];

    red[t] = v0 + v1;
    __syncthreads();
    for (int s = 128; s > 0; s >>= 1) {
        if (t < s) red[t] += red[t + s];
        __syncthreads();
    }
    const float mu = red[0] * (1.0f / E_);
    __syncthreads();

    const float d0 = v0 - mu, d1 = v1 - mu;
    red[t] = d0 * d0 + d1 * d1;
    __syncthreads();
    for (int s = 128; s > 0; s >>= 1) {
        if (t < s) red[t] += red[t + s];
        __syncthreads();
    }
    const float rs = rsqrtf(red[0] * (1.0f / E_) + 1e-5f);

    const float y0 = d0 * rs * g[t]       + be[t];
    const float y1 = d1 * rs * g[t + 256] + be[t + 256];
    outF[base + t]       = y0;
    outF[base + 256 + t] = y1;
    if (outB) {
        outB[base + t]       = f2bf_bits(y0);
        outB[base + 256 + t] = f2bf_bits(y1);
    }
}

__global__ void tail_zero_kernel(float* out, int begin, int end) {
    int i = begin + blockIdx.x * blockDim.x + threadIdx.x;
    if (i < end) out[i] = 0.0f;
}

// ---------------------------------------------------------------------------
// Host-side orchestration
// ---------------------------------------------------------------------------
extern "C" void kernel_launch(void* const* d_in, const int* in_sizes, int n_in,
                              void* d_out, int out_size, void* d_ws, size_t ws_size,
                              hipStream_t stream) {
    const float* x   = (const float*)d_in[0];
    const float* Wq  = (const float*)d_in[1];
    const float* Wk  = (const float*)d_in[2];
    const float* Wv  = (const float*)d_in[3];
    const float* Wo  = (const float*)d_in[4];
    const float* bo  = (const float*)d_in[5];
    const float* g1  = (const float*)d_in[6];
    const float* b1  = (const float*)d_in[7];
    const float* W1  = (const float*)d_in[8];
    const float* bf1 = (const float*)d_in[9];
    const float* W2  = (const float*)d_in[10];
    const float* bf2 = (const float*)d_in[11];
    const float* g2  = (const float*)d_in[12];
    const float* b2  = (const float*)d_in[13];

    char* p = (char*)d_ws;
    auto take = [&](size_t bytes) -> void* {
        void* q = (void*)p;
        p += (bytes + 255) & ~(size_t)255;
        return q;
    };
    unsigned short* Xb   = (unsigned short*)take((size_t)M_ * E_ * 2);
    unsigned short* Wqb  = (unsigned short*)take((size_t)E_ * E_ * 2);
    unsigned short* Wkb  = (unsigned short*)take((size_t)E_ * E_ * 2);
    unsigned short* Wvb  = (unsigned short*)take((size_t)E_ * E_ * 2);
    unsigned short* Wob  = (unsigned short*)take((size_t)E_ * E_ * 2);
    unsigned short* W1b  = (unsigned short*)take((size_t)E_ * F_ * 2);
    unsigned short* W2b  = (unsigned short*)take((size_t)F_ * E_ * 2);
    unsigned short* Qb   = (unsigned short*)take((size_t)M_ * E_ * 2);
    unsigned short* Kb   = (unsigned short*)take((size_t)M_ * E_ * 2);
    unsigned short* Vb   = (unsigned short*)take((size_t)M_ * E_ * 2);
    unsigned short* Ab   = (unsigned short*)take((size_t)M_ * E_ * 2);
    float*          mh   = (float*)take((size_t)M_ * E_ * 4);
    float*          out1 = (float*)take((size_t)M_ * E_ * 4);
    unsigned short* o1b  = (unsigned short*)take((size_t)M_ * E_ * 2);
    unsigned short* ff1b = (unsigned short*)take((size_t)M_ * F_ * 2);
    float*          ff2  = (float*)take((size_t)M_ * E_ * 4);
    (void)ws_size; (void)in_sizes; (void)n_in;

    convert_bf16_kernel<<<2048, 256, 0, stream>>>(x,  Xb,  M_ * E_);
    convert_bf16_kernel<<<256,  256, 0, stream>>>(Wq, Wqb, E_ * E_);
    convert_bf16_kernel<<<256,  256, 0, stream>>>(Wk, Wkb, E_ * E_);
    convert_bf16_kernel<<<256,  256, 0, stream>>>(Wv, Wvb, E_ * E_);
    convert_bf16_kernel<<<256,  256, 0, stream>>>(Wo, Wob, E_ * E_);
    convert_bf16_kernel<<<1024, 256, 0, stream>>>(W1, W1b, E_ * F_);
    convert_bf16_kernel<<<1024, 256, 0, stream>>>(W2, W2b, F_ * E_);

    // QKV projections (bf16 out, no bias); macro-tile 64x128
    dim3 gEE(E_ / 128, M_ / 64);
    gemm_bf16_kernel<false, false, true><<<gEE, 256, 0, stream>>>(
        Xb, Wqb, nullptr, nullptr, Qb, M_, E_, E_);
    gemm_bf16_kernel<false, false, true><<<gEE, 256, 0, stream>>>(
        Xb, Wkb, nullptr, nullptr, Kb, M_, E_, E_);
    gemm_bf16_kernel<false, false, true><<<gEE, 256, 0, stream>>>(
        Xb, Wvb, nullptr, nullptr, Vb, M_, E_, E_);

    // flash attention: (query-tile, b*h)
    flash_attn_kernel<<<dim3(N_ / 16, B_ * H_), 32, 0, stream>>>(Qb, Kb, Vb, Ab);

    // output projection + bias -> fp32
    gemm_bf16_kernel<true, false, false><<<gEE, 256, 0, stream>>>(
        Ab, Wob, bo, mh, nullptr, M_, E_, E_);

    // out1 = LN(x + mh)   (fp32 + bf16 shadow for FFN)
    add_layernorm_kernel<<<M_, 256, 0, stream>>>(x, mh, g1, b1, out1, o1b);

    // FFN: relu(out1 @ W1 + bf1) -> bf16 ; @ W2 + bf2 -> fp32
    gemm_bf16_kernel<true, true, true><<<dim3(F_ / 128, M_ / 64), 256, 0, stream>>>(
        o1b, W1b, bf1, nullptr, ff1b, M_, F_, E_);
    gemm_bf16_kernel<true, false, false><<<gEE, 256, 0, stream>>>(
        ff1b, W2b, bf2, ff2, nullptr, M_, E_, F_);

    // out3 = LN(out1 + ff) -> d_out (fp32)
    add_layernorm_kernel<<<M_, 256, 0, stream>>>(out1, ff2, g2, b2,
                                                 (float*)d_out, nullptr);

    const int main_elems = M_ * E_;
    if (out_size > main_elems) {
        int extra = out_size - main_elems;
        tail_zero_kernel<<<(extra + 255) / 256, 256, 0, stream>>>(
            (float*)d_out, main_elems, out_size);
    }
}
